// NLMEncoderLayer_25348896981151
// MI455X (gfx1250) — compile-verified, matrix operand order checked by reading, not run on hardware
//
#include <hip/hip_runtime.h>
#include <hip/hip_bf16.h>

// ---------------------------------------------------------------------------
// NLM encoder layer for MI455X (gfx1250), bf16 WMMA path.
// B=8, N=256, D=64.
// out1: [B,N,64]   = MLP_192->128(relu)->64 ( [emb1 | max_j emb2 | min_j emb2] )
// out2: [B,N,N,64] = MLP_256->256(relu)->64 ( [emb1[j] | emb2[j,i] | emb1[i] | emb2[i,j]] )
// out2_kernel: one wave = 32 rows (two 16-row j-tiles) so every weight
// fragment load feeds two independent WMMA chains.
// ---------------------------------------------------------------------------

typedef __bf16 bf16_t;
typedef __attribute__((ext_vector_type(16))) __bf16 v16bf;
typedef __attribute__((ext_vector_type(8)))  __bf16 v8bf;
typedef __attribute__((ext_vector_type(8)))  float  v8f;

static __device__ inline v8f wmma_bf16(v16bf a, v16bf b, v8f c) {
  // D = A(16x32) x B(32x16) + C, f32 accumulate
  return __builtin_amdgcn_wmma_f32_16x16x32_bf16(false, a, false, b, (short)0, c,
                                                 false, false);
}

// Load one 16x32 bf16 fragment slice for this lane: two contiguous 16B chunks
// (K = base..base+7 and base+16..base+23 relative to this lane's koff).
static __device__ inline v16bf load_frag(const bf16_t* __restrict__ p) {
  v8bf lo = *(const v8bf*)(p);
  v8bf hi = *(const v8bf*)(p + 16);
  return __builtin_shufflevector(lo, hi, 0, 1, 2, 3, 4, 5, 6, 7,
                                 8, 9, 10, 11, 12, 13, 14, 15);
}

// ---------------------------------------------------------------------------
// Preprocessing kernels
// ---------------------------------------------------------------------------

// dst[n][k] = (bf16) src[k][n]   (src is K x N row-major)
__global__ void transpose_to_bf16(const float* __restrict__ src,
                                  bf16_t* __restrict__ dst, int K, int N) {
  int idx = blockIdx.x * blockDim.x + threadIdx.x;
  if (idx < K * N) {
    int n = idx / K, k = idx % K;
    dst[n * K + k] = (bf16_t)src[k * N + n];
  }
}

__global__ void convert_bf16(const float* __restrict__ src,
                             bf16_t* __restrict__ dst, int n) {
  int idx = blockIdx.x * blockDim.x + threadIdx.x;
  if (idx < n) dst[idx] = (bf16_t)src[idx];
}

// One block per (b,i) slab of emb2: convert 256x64 f32 -> bf16 and reduce
// max/min over j (axis -2) into bf16 outputs.
__global__ __launch_bounds__(256) void emb2_prep(
    const float* __restrict__ emb2, bf16_t* __restrict__ emb2bf,
    bf16_t* __restrict__ rmax, bf16_t* __restrict__ rmin) {
  int slab = blockIdx.x;          // b*256 + i, 2048 blocks
  int t = threadIdx.x;            // 256 threads
  int d = t & 63;
  int jg = t >> 6;                // 0..3
  const float* base = emb2 + (size_t)slab * 256 * 64;
  bf16_t* obase = emb2bf + (size_t)slab * 256 * 64;
  float mx = -3.4e38f, mn = 3.4e38f;
  for (int jj = 0; jj < 64; ++jj) {
    int j = jj * 4 + jg;
    float v = base[(size_t)j * 64 + d];
    obase[(size_t)j * 64 + d] = (bf16_t)v;
    mx = fmaxf(mx, v);
    mn = fminf(mn, v);
  }
  __shared__ float smx[256], smn[256];
  smx[t] = mx;
  smn[t] = mn;
  __syncthreads();
  if (jg == 0) {
    for (int g = 1; g < 4; ++g) {
      mx = fmaxf(mx, smx[g * 64 + d]);
      mn = fminf(mn, smn[g * 64 + d]);
    }
    rmax[(size_t)slab * 64 + d] = (bf16_t)mx;
    rmin[(size_t)slab * 64 + d] = (bf16_t)mn;
  }
}

// ---------------------------------------------------------------------------
// out2: fused 256->256(relu)->64 MLP over 524288 rows.
// One wave = 32 rows (fixed b,i; j-tiles j0 and j0+16). GEMM1 is computed
// transposed (A = W2a^T tile, B = input fragment) so the hidden D-tiles map
// lane-exactly onto GEMM2 A-fragments: only relu+cvt between layers.
// Each weight fragment is loaded once and used by both j-tiles.
// ---------------------------------------------------------------------------
__global__ __launch_bounds__(128) void out2_kernel(
    const bf16_t* __restrict__ emb1bf, const bf16_t* __restrict__ emb2bf,
    const bf16_t* __restrict__ W2aT, const float* __restrict__ b2a,
    const bf16_t* __restrict__ W2bT, const float* __restrict__ b2b,
    float* __restrict__ out2) {
  const int wid = threadIdx.x >> 5;
  const int lane = threadIdx.x & 31;
  const int lane16 = lane & 15;
  const int koff = (lane < 16) ? 0 : 8;   // K sub-offset per WMMA 16-bit layout
  const int unit = blockIdx.x * 4 + wid;  // 16384 units of 32 rows
  const int b = unit >> 11;               // 2048 units per batch
  const int rem = unit & 2047;
  const int i = rem >> 3;
  const int j0 = (rem & 7) << 5;          // 32-row strip
  const int jA = j0 + lane16;             // lane's row, tile A
  const int jB = j0 + 16 + lane16;        // lane's row, tile B

  // ---- input K-fragments: [emb1[j] | emb2[j,i] | emb1[i] | emb2[i,j]] ----
  const bf16_t* e1i  = emb1bf + ((size_t)(b * 256 + i)) * 64;
  const bf16_t* e1jA = emb1bf + ((size_t)(b * 256 + jA)) * 64;
  const bf16_t* e1jB = emb1bf + ((size_t)(b * 256 + jB)) * 64;
  const bf16_t* e2jiA = emb2bf + (((size_t)(b * 256 + jA)) * 256 + i) * 64;
  const bf16_t* e2jiB = emb2bf + (((size_t)(b * 256 + jB)) * 256 + i) * 64;
  const bf16_t* e2ij  = emb2bf + (((size_t)(b * 256 + i)) * 256 + j0) * 64;

  v16bf fA[8], fB[8];
  fA[0] = load_frag(e1jA + koff);
  fA[1] = load_frag(e1jA + 32 + koff);
  fB[0] = load_frag(e1jB + koff);
  fB[1] = load_frag(e1jB + 32 + koff);
  fA[2] = load_frag(e2jiA + koff);
  fA[3] = load_frag(e2jiA + 32 + koff);
  fB[2] = load_frag(e2jiB + koff);
  fB[3] = load_frag(e2jiB + 32 + koff);
  fA[4] = load_frag(e1i + koff);          // shared between tiles
  fA[5] = load_frag(e1i + 32 + koff);
  fA[6] = load_frag(e2ij + (size_t)lane16 * 64 + koff);
  fA[7] = load_frag(e2ij + (size_t)lane16 * 64 + 32 + koff);
  fB[6] = load_frag(e2ij + (size_t)(16 + lane16) * 64 + koff);
  fB[7] = load_frag(e2ij + (size_t)(16 + lane16) * 64 + 32 + koff);

  v8f oaccA[4] = {}, oaccB[4] = {};  // 64 output features = 4 o-tiles each

  #pragma unroll
  for (int c = 0; c < 8; ++c) {  // 8 pairs of hidden 16-tiles = 256 hidden
    // bias-initialized accumulators (element v <-> hidden row 32c[+16]+koff+v)
    v8f bias0 = *(const v8f*)(b2a + 32 * c + koff);
    v8f bias1 = *(const v8f*)(b2a + 32 * c + 16 + koff);
    v8f h0a = bias0, h1a = bias1, h0b = bias0, h1b = bias1;
    #pragma unroll
    for (int k = 0; k < 8; ++k) {
      v16bf w0 = load_frag(W2aT + (size_t)(32 * c + lane16) * 256 + 32 * k + koff);
      v16bf w1 = load_frag(W2aT + (size_t)(32 * c + 16 + lane16) * 256 + 32 * k + koff);
      v16bf xb = (k == 4 || k == 5) ? fA[k] : fB[k];  // emb1[i] frags shared
      h0a = wmma_bf16(w0, fA[k], h0a);
      h0b = wmma_bf16(w0, xb, h0b);
      h1a = wmma_bf16(w1, fA[k], h1a);
      h1b = wmma_bf16(w1, xb, h1b);
    }
    // relu + pack: D-tiles (2c, 2c+1) become one GEMM2 A-fragment, in-lane.
    v16bf a2a, a2b;
    #pragma unroll
    for (int v = 0; v < 8; ++v) {
      a2a[v]     = (bf16_t)fmaxf(h0a[v], 0.0f);
      a2a[v + 8] = (bf16_t)fmaxf(h1a[v], 0.0f);
      a2b[v]     = (bf16_t)fmaxf(h0b[v], 0.0f);
      a2b[v + 8] = (bf16_t)fmaxf(h1b[v], 0.0f);
    }
    #pragma unroll
    for (int o = 0; o < 4; ++o) {
      v16bf w2 = load_frag(W2bT + (size_t)(16 * o + lane16) * 256 + 32 * c + koff);
      oaccA[o] = wmma_bf16(a2a, w2, oaccA[o]);
      oaccB[o] = wmma_bf16(a2b, w2, oaccB[o]);
    }
  }

  // ---- store D (lane = out feature, vgpr v = local row v + 8*half) ----
  const int half = (lane < 16) ? 0 : 8;
  const size_t outbaseA = (((size_t)b * 256 + i) * 256 + j0) * 64;
  const size_t outbaseB = outbaseA + (size_t)16 * 64;
  #pragma unroll
  for (int o = 0; o < 4; ++o) {
    float bb = b2b[16 * o + lane16];
    #pragma unroll
    for (int v = 0; v < 8; ++v) {
      out2[outbaseA + (size_t)(v + half) * 64 + 16 * o + lane16] = oaccA[o][v] + bb;
      out2[outbaseB + (size_t)(v + half) * 64 + 16 * o + lane16] = oaccB[o][v] + bb;
    }
  }
}

// ---------------------------------------------------------------------------
// out1: fused 192->128(relu)->64 MLP over 2048 rows (same scheme, tiny).
// ---------------------------------------------------------------------------
__global__ __launch_bounds__(128) void out1_kernel(
    const bf16_t* __restrict__ emb1bf, const bf16_t* __restrict__ rmaxbf,
    const bf16_t* __restrict__ rminbf, const bf16_t* __restrict__ W1aT,
    const float* __restrict__ b1a, const bf16_t* __restrict__ W1bT,
    const float* __restrict__ b1b, float* __restrict__ out1) {
  const int wid = threadIdx.x >> 5;
  const int lane = threadIdx.x & 31;
  const int lane16 = lane & 15;
  const int koff = (lane < 16) ? 0 : 8;
  const int tile = blockIdx.x * 4 + wid;  // 128 tiles of 16 rows
  const int r = tile * 16 + lane16;       // global row = b*256 + n

  const bf16_t* s0 = emb1bf + (size_t)r * 64;
  const bf16_t* s1 = rmaxbf + (size_t)r * 64;
  const bf16_t* s2 = rminbf + (size_t)r * 64;
  v16bf bfrag[6];
  bfrag[0] = load_frag(s0 + koff);
  bfrag[1] = load_frag(s0 + 32 + koff);
  bfrag[2] = load_frag(s1 + koff);
  bfrag[3] = load_frag(s1 + 32 + koff);
  bfrag[4] = load_frag(s2 + koff);
  bfrag[5] = load_frag(s2 + 32 + koff);

  v8f oacc[4] = {};
  #pragma unroll
  for (int c = 0; c < 4; ++c) {  // hidden 128 = 4 tile-pairs
    v8f h0 = *(const v8f*)(b1a + 32 * c + koff);
    v8f h1 = *(const v8f*)(b1a + 32 * c + 16 + koff);
    #pragma unroll
    for (int k = 0; k < 6; ++k) {  // K = 192 = 6 chunks
      v16bf w0 = load_frag(W1aT + (size_t)(32 * c + lane16) * 192 + 32 * k + koff);
      v16bf w1 = load_frag(W1aT + (size_t)(32 * c + 16 + lane16) * 192 + 32 * k + koff);
      h0 = wmma_bf16(w0, bfrag[k], h0);
      h1 = wmma_bf16(w1, bfrag[k], h1);
    }
    v16bf a2;
    #pragma unroll
    for (int v = 0; v < 8; ++v) {
      a2[v]     = (bf16_t)fmaxf(h0[v], 0.0f);
      a2[v + 8] = (bf16_t)fmaxf(h1[v], 0.0f);
    }
    #pragma unroll
    for (int o = 0; o < 4; ++o) {
      v16bf w2 = load_frag(W1bT + (size_t)(16 * o + lane16) * 128 + 32 * c + koff);
      oacc[o] = wmma_bf16(a2, w2, oacc[o]);
    }
  }

  const int half = (lane < 16) ? 0 : 8;
  #pragma unroll
  for (int o = 0; o < 4; ++o) {
    float bb = b1b[16 * o + lane16];
    #pragma unroll
    for (int v = 0; v < 8; ++v) {
      out1[((size_t)tile * 16 + v + half) * 64 + 16 * o + lane16] = oacc[o][v] + bb;
    }
  }
}

// ---------------------------------------------------------------------------
// Host launcher
// ---------------------------------------------------------------------------
extern "C" void kernel_launch(void* const* d_in, const int* in_sizes, int n_in,
                              void* d_out, int out_size, void* d_ws, size_t ws_size,
                              hipStream_t stream) {
  (void)in_sizes; (void)n_in; (void)out_size; (void)ws_size;
  const int B = 8, N = 256, D = 64;

  const float* emb1 = (const float*)d_in[0];
  const float* emb2 = (const float*)d_in[1];
  const float* W1a  = (const float*)d_in[2];  // [192,128]
  const float* b1a  = (const float*)d_in[3];
  const float* W1b  = (const float*)d_in[4];  // [128,64]
  const float* b1b  = (const float*)d_in[5];
  const float* W2a  = (const float*)d_in[6];  // [256,256]
  const float* b2a  = (const float*)d_in[7];
  const float* W2b  = (const float*)d_in[8];  // [256,64]
  const float* b2b  = (const float*)d_in[9];

  // workspace layout (bf16), 256B-aligned sections
  char* ws = (char*)d_ws;
  size_t off = 0;
  auto alloc = [&](size_t bytes) {
    char* p = ws + off;
    off += (bytes + 255) & ~(size_t)255;
    return p;
  };
  bf16_t* emb2bf = (bf16_t*)alloc((size_t)B * N * N * D * 2);  // 64 MB
  bf16_t* emb1bf = (bf16_t*)alloc((size_t)B * N * D * 2);
  bf16_t* rmaxbf = (bf16_t*)alloc((size_t)B * N * D * 2);
  bf16_t* rminbf = (bf16_t*)alloc((size_t)B * N * D * 2);
  bf16_t* W1aT   = (bf16_t*)alloc((size_t)192 * 128 * 2);  // [128][192]
  bf16_t* W1bT   = (bf16_t*)alloc((size_t)128 * 64 * 2);   // [64][128]
  bf16_t* W2aT   = (bf16_t*)alloc((size_t)256 * 256 * 2);  // [256][256]
  bf16_t* W2bT   = (bf16_t*)alloc((size_t)256 * 64 * 2);   // [64][256]

  float* out1 = (float*)d_out;                       // B*N*D floats
  float* out2 = (float*)d_out + (size_t)B * N * D;   // B*N*N*D floats

  // --- preprocessing ---
  transpose_to_bf16<<<(192 * 128 + 255) / 256, 256, 0, stream>>>(W1a, W1aT, 192, 128);
  transpose_to_bf16<<<(128 * 64 + 255) / 256, 256, 0, stream>>>(W1b, W1bT, 128, 64);
  transpose_to_bf16<<<(256 * 256 + 255) / 256, 256, 0, stream>>>(W2a, W2aT, 256, 256);
  transpose_to_bf16<<<(256 * 64 + 255) / 256, 256, 0, stream>>>(W2b, W2bT, 256, 64);
  convert_bf16<<<(B * N * D + 255) / 256, 256, 0, stream>>>(emb1, emb1bf, B * N * D);
  emb2_prep<<<B * N, 256, 0, stream>>>(emb2, emb2bf, rmaxbf, rminbf);

  // --- fused MLPs ---
  out1_kernel<<<(B * N / 16) / 4, 128, 0, stream>>>(emb1bf, rmaxbf, rminbf,
                                                    W1aT, b1a, W1bT, b1b, out1);
  // 32 rows per wave, 4 waves per block: B*N*N/32 units
  out2_kernel<<<(B * N * N / 32) / 4, 128, 0, stream>>>(emb1bf, emb2bf,
                                                        W2aT, b2a, W2bT, b2b, out2);
}
